// DynamicAttention_77197742178888
// MI455X (gfx1250) — matrix operand order repaired and checked by reading, not compiled, not run on hardware
//
#include <hip/hip_runtime.h>
#include <hip/hip_bf16.h>
#include <math.h>

typedef __attribute__((ext_vector_type(16))) _Float16 v16h;
typedef __attribute__((ext_vector_type(8)))  _Float16 v8h;
typedef __attribute__((ext_vector_type(4)))  _Float16 v4h;
typedef __attribute__((ext_vector_type(8)))  float    v8f;

#define EPSV 1e-10f
#define PADK 40   // padded K-stride (halfs): row start = n*80B -> 16B aligned, bank-conflict free

union F16F { v16h v; v8h h8[2]; _Float16 h[16]; };
union F32A { v8f v; float f[8]; };

__device__ __forceinline__ float delu_f(float x) {
  return 10.0f * fmaxf(x, 0.0f) + __expf(10.0f * fminf(x, 0.0f));
}

__device__ __forceinline__ v8f wmma_f16(const F16F& a, const F16F& b, v8f c) {
  return __builtin_amdgcn_wmma_f32_16x16x32_f16(false, a.v, false, b.v,
                                                (short)0, c, false, false);
}

// ---------------------------------------------------------------------------
// FAST 1x1 conv / per-batch GEMM: requires O%32==0, C%32==0, Npix%128==0,
// bias non-null. 256 thr = 8 waves. Block tile: 32(O) x 128(N); each wave does
// 2 WMMAs per K-step sharing its A fragment. K-contiguous LDS -> b128 frags.
// ---------------------------------------------------------------------------
template <int ACT, bool CS>
__global__ __launch_bounds__(256)
void conv1x1_fast(const float* __restrict__ X, int XC,
                  const float* __restrict__ Wt, long wbstride, int C,
                  const float* __restrict__ bias,
                  const float* __restrict__ colscale,
                  float* __restrict__ Y, int YC, int O, int Npix)
{
  __shared__ _Float16 lA[32 * PADK];    // [m][k]
  __shared__ _Float16 lB[128 * PADK];   // [n][k]
  const int b  = blockIdx.z;
  const int o0 = blockIdx.y << 5;
  const int n0 = blockIdx.x << 7;
  const int t = threadIdx.x, wave = t >> 5, lane = t & 31;
  const int ow = wave & 1, nw = wave >> 1;
  const float* Wb = Wt + (long)b * wbstride;
  const float* Xb = X + (long)b * XC * (long)Npix;
  v8f acc0 = {}, acc1 = {};

  for (int k0 = 0; k0 < C; k0 += 32) {
    { // stage A: 32 rows x 32 k; thread: one float4 along k
      const int m = t >> 3, kc = (t & 7) << 2;
      const float* wp = &Wb[(long)(o0 + m) * C + k0 + kc];
      const float4 w4 = *reinterpret_cast<const float4*>(wp);
      if (k0 + 32 < C) __builtin_prefetch(wp + 32, 0, 3);   // global_prefetch_b8
      v4h h; h.x = (_Float16)w4.x; h.y = (_Float16)w4.y;
             h.z = (_Float16)w4.z; h.w = (_Float16)w4.w;
      *reinterpret_cast<v4h*>(&lA[m * PADK + kc]) = h;
    }
    { // stage B: 128 n x 32 k; thread: 16 k-values for one n (coalesced over n)
      const int nn = t & 127, kg = (t >> 7) << 4;
      const float* xp = &Xb[(long)(k0 + kg) * Npix + n0 + nn];
      if (k0 + 32 < C) __builtin_prefetch(xp + (long)32 * Npix, 0, 3);
      v8h h0, h1;
#pragma unroll
      for (int j = 0; j < 8; ++j) h0[j] = (_Float16)xp[(long)j * Npix];
#pragma unroll
      for (int j = 0; j < 8; ++j) h1[j] = (_Float16)xp[(long)(8 + j) * Npix];
      *reinterpret_cast<v8h*>(&lB[nn * PADK + kg]) = h0;
      *reinterpret_cast<v8h*>(&lB[nn * PADK + kg + 8]) = h1;
    }
    __syncthreads();

    F16F af, bf0, bf1;
    const int ka = (lane < 16) ? 0 : 8;
    const int kb = (lane < 16) ? 0 : 8;
    const _Float16* pa = &lA[((ow << 4) + (lane & 15)) * PADK + ka];
    af.h8[0] = *reinterpret_cast<const v8h*>(pa);
    af.h8[1] = *reinterpret_cast<const v8h*>(pa + 16);
    const _Float16* pb0 = &lB[((nw << 5) + (lane & 15)) * PADK + kb];
    bf0.h8[0] = *reinterpret_cast<const v8h*>(pb0);
    bf0.h8[1] = *reinterpret_cast<const v8h*>(pb0 + 16);
    const _Float16* pb1 = &lB[((nw << 5) + 16 + (lane & 15)) * PADK + kb];
    bf1.h8[0] = *reinterpret_cast<const v8h*>(pb1);
    bf1.h8[1] = *reinterpret_cast<const v8h*>(pb1 + 16);
    acc0 = wmma_f16(af, bf0, acc0);
    acc1 = wmma_f16(af, bf1, acc1);
    __syncthreads();
  }

  const int mb = (lane < 16) ? 0 : 8;
  const int ncol = n0 + (nw << 5) + (lane & 15);
  float cs = 1.0f;
  if (CS) cs = colscale[(long)b * Npix + ncol];
  F32A a0, a1; a0.v = acc0; a1.v = acc1;
#pragma unroll
  for (int j = 0; j < 8; ++j) {
    const int o = o0 + (ow << 4) + mb + j;
    const float bz = bias[o];
    float u0 = a0.f[j] + bz, u1 = a1.f[j] + bz;
    if (CS) { u0 *= cs; u1 *= cs; }
    if (ACT) { u0 = delu_f(u0); u1 = delu_f(u1); }
    float* yp = &Y[((long)b * YC + o) * Npix + ncol];
    yp[0] = u0; yp[16] = u1;
  }
}

// ---------------------------------------------------------------------------
// 3x3 conv (pad=1), 9 accumulated WMMA GEMMs. 256 thr = 8 waves.
// Block: one image row y, 32(O) x 64(pixels). Halo/out-of-range rows of the
// input tile are zeroed once; in-loop staging skips out-of-range rows & is coalesced.
// ---------------------------------------------------------------------------
__global__ __launch_bounds__(256)
void conv3x3_wmma(const float* __restrict__ X,   // (B,C,64,64)
                  const float* __restrict__ Wt,  // (O,C,3,3)
                  const float* __restrict__ bias,
                  float* __restrict__ Y, int C, int O)
{
  const int WW = 64, Npix = 4096;
  __shared__ _Float16 lA[9 * 32 * PADK];     // [tap][m][k]
  __shared__ _Float16 lB[3 * 66 * PADK];     // [dy][x+1][k]
  const int b = blockIdx.z, y = blockIdx.y, o0 = blockIdx.x << 5;
  const int t = threadIdx.x, wave = t >> 5, lane = t & 31;
  const int ow = wave & 1, nwv = wave >> 1;
  v8f acc = {};

  for (int i = t; i < 3 * 66 * PADK; i += 256) lB[i] = (_Float16)0.0f;
  __syncthreads();

  for (int k0 = 0; k0 < C; k0 += 32) {
    // stage A: 32 o x 32 c x 9 taps (taps contiguous in global memory)
    for (int i = t; i < 9 * 32 * 32; i += 256) {
      const int tap = i % 9, rem = i / 9, kk = rem & 31, m = rem >> 5;
      const float* wp = &Wt[((long)(o0 + m) * C + k0 + kk) * 9 + tap];
      if (k0 + 32 < C) __builtin_prefetch(wp + 32 * 9, 0, 3);
      lA[(tap * 32 + m) * PADK + kk] = (_Float16)(*wp);
    }
    // stage B interior: 3 rows x 32 c x 64 cols; dy-validity is uniform per iter
    for (int i = t; i < 3 * 32 * 64; i += 256) {
      const int xx = i & 63, rem = i >> 6, kk = rem & 31, dy = rem >> 5;
      const int yy = y + dy - 1;
      if (yy >= 0 && yy < 64) {
        const float* xp = &X[((long)b * C + (k0 + kk)) * Npix + yy * WW + xx];
        if (k0 + 32 < C) __builtin_prefetch(xp + (long)32 * Npix, 0, 3);
        lB[(dy * 66 + xx + 1) * PADK + kk] = (_Float16)(*xp);
      }
    }
    __syncthreads();

    const int ka = (lane < 16) ? 0 : 8;
    const int kb = (lane < 16) ? 0 : 8;
    const int ncol = (nwv << 4) + (lane & 15);      // 0..63
#pragma unroll
    for (int dy = 0; dy < 3; ++dy) {
#pragma unroll
      for (int dx = 0; dx < 3; ++dx) {
        F16F af, bf;
        const _Float16* pa =
            &lA[(((dy * 3 + dx) * 32) + (ow << 4) + (lane & 15)) * PADK + ka];
        af.h8[0] = *reinterpret_cast<const v8h*>(pa);
        af.h8[1] = *reinterpret_cast<const v8h*>(pa + 16);
        const _Float16* pb = &lB[(dy * 66 + ncol + dx) * PADK + kb];
        bf.h8[0] = *reinterpret_cast<const v8h*>(pb);
        bf.h8[1] = *reinterpret_cast<const v8h*>(pb + 16);
        acc = wmma_f16(af, bf, acc);
      }
    }
    __syncthreads();
  }

  const int mb = (lane < 16) ? 0 : 8;
  const int ncol = (nwv << 4) + (lane & 15);
  F32A ac; ac.v = acc;
#pragma unroll
  for (int j = 0; j < 8; ++j) {
    const int o = o0 + (ow << 4) + mb + j;
    Y[((long)b * O + o) * Npix + y * WW + ncol] = ac.f[j] + bias[o];
  }
}

// ---------------------------------------------------------------------------
// Zero-pad a (srows x cols) tensor into (drows x cols); rows >= srows -> 0.
// ---------------------------------------------------------------------------
__global__ __launch_bounds__(256)
void pad2d_kernel(const float* __restrict__ src, float* __restrict__ dst,
                  int srows, int cols, long total) {
  const long i = (long)blockIdx.x * 256 + threadIdx.x;
  if (i >= total) return;
  const long r = i / cols;
  dst[i] = (r < srows) ? src[i - (i / cols) * cols + r * cols] : 0.0f;
}

// ---------------------------------------------------------------------------
// Small reductions / elementwise (<1% of FLOPs).  q/k live in padded (B,64,N)
// layout whose rows 40..63 are exactly zero.
// ---------------------------------------------------------------------------
__global__ __launch_bounds__(256)
void rowsum_kernel(const float* __restrict__ X, float* __restrict__ out, int Npix) {
  __shared__ float s[256];
  const long base = (long)blockIdx.x * Npix;
  float a = 0.0f;
  for (int n = threadIdx.x; n < Npix; n += 256) a += X[base + n];
  s[threadIdx.x] = a; __syncthreads();
  for (int st = 128; st > 0; st >>= 1) {
    if (threadIdx.x < st) s[threadIdx.x] += s[threadIdx.x + st];
    __syncthreads();
  }
  if (threadIdx.x == 0) out[blockIdx.x] = s[0];
}

// kvt[b, c*64+m] = sum_n K[b,m,n] * V[b,c,n]; m in [0,64) (rows >=40 are zero)
__global__ __launch_bounds__(256)
void kv_kernel(const float* __restrict__ K, const float* __restrict__ V,
               float* __restrict__ kvt, int Npix) {
  __shared__ float s[256];
  const int c = blockIdx.x, m = blockIdx.y, b = blockIdx.z;
  const float* kp = K + ((long)b * 64 + m) * Npix;
  const float* vp = V + ((long)b * 320 + c) * Npix;
  float a = 0.0f;
  for (int n = threadIdx.x; n < Npix; n += 256) a += kp[n] * vp[n];
  s[threadIdx.x] = a; __syncthreads();
  for (int st = 128; st > 0; st >>= 1) {
    if (threadIdx.x < st) s[threadIdx.x] += s[threadIdx.x + st];
    __syncthreads();
  }
  if (threadIdx.x == 0) kvt[((long)b * 320 + c) * 64 + m] = s[0];
}

__global__ __launch_bounds__(256)
void colsum_kernel(const float* __restrict__ X, float* __restrict__ out,
                   int C, int Npix) {
  const int b = blockIdx.y;
  const int n = blockIdx.x * 256 + threadIdx.x;
  const float* Xb = X + (long)b * C * Npix;
  float a = 0.0f;
  for (int c = 0; c < C; ++c) a += Xb[(long)c * Npix + n];
  out[(long)b * Npix + n] = a;
}

__global__ __launch_bounds__(256)
void dot_rows_kernel(const float* __restrict__ A, const float* __restrict__ X,
                     float* __restrict__ out, int C, int Npix) {
  __shared__ float s[256];
  const int c = blockIdx.x, b = blockIdx.y;
  const float* ap = A + (long)b * Npix;
  const float* xp = X + ((long)b * C + c) * Npix;
  float a = 0.0f;
  for (int n = threadIdx.x; n < Npix; n += 256) a += ap[n] * xp[n];
  s[threadIdx.x] = a; __syncthreads();
  for (int st = 128; st > 0; st >>= 1) {
    if (threadIdx.x < st) s[threadIdx.x] += s[threadIdx.x + st];
    __syncthreads();
  }
  if (threadIdx.x == 0) out[(long)b * C + c] = s[0];
}

// norm[b,n]  = 1 / sum_m q[b,m,n]*(ksum[b,m]+EPS)   (m in [0,64); rows>=40 zero)
// norml[b,n] = 1 / (sum_c es[b,c]*kl[b,c,n] + EPS)
__global__ __launch_bounds__(256)
void norms_kernel(const float* __restrict__ Q, const float* __restrict__ ksum,
                  const float* __restrict__ ES, const float* __restrict__ KL,
                  float* __restrict__ norm, float* __restrict__ norml, int Npix) {
  const int b = blockIdx.y;
  const int n = blockIdx.x * 256 + threadIdx.x;
  float a = 0.0f;
  for (int m = 0; m < 64; ++m)
    a += Q[((long)b * 64 + m) * Npix + n] * (ksum[b * 64 + m] + EPSV);
  norm[(long)b * Npix + n] = 1.0f / a;
  float l = 0.0f;
  for (int c = 0; c < 320; ++c)
    l += ES[b * 320 + c] * KL[((long)b * 320 + c) * Npix + n];
  norml[(long)b * Npix + n] = 1.0f / (l + EPSV);
}

__global__ __launch_bounds__(256)
void compose_kernel(float* __restrict__ xb,
                    const float* __restrict__ wv, const float* __restrict__ d1,
                    const float* __restrict__ d3, const float* __restrict__ kl,
                    const float* __restrict__ vl, const float* __restrict__ d1l,
                    const float* __restrict__ d3l, const float* __restrict__ es,
                    const float* __restrict__ norml,
                    const float* __restrict__ ag, const float* __restrict__ lg,
                    int C, long Npix, long total) {
  const long idx = (long)blockIdx.x * 256 + threadIdx.x;
  if (idx >= total) return;
  const long n = idx % Npix;
  const long c = (idx / Npix) % C;
  const long b = idx / (Npix * C);
  const float x = xb[idx];
  const float attn = x + ag[0] * (wv[idx] + 2.0f * (d1[idx] + d3[idx]));
  const float attl = es[b * C + c] * kl[idx] * vl[idx] * norml[b * Npix + n];
  const float lin  = x + lg[0] * (attl + d1l[idx] + d3l[idx]);
  xb[idx] = 0.7f * attn + 0.3f * lin + x;
}

// ---------------------------------------------------------------------------
static inline void c1(const float* X, int XC, const float* W, long wbs, int C,
                      const float* bias, const float* cs, float* Y, int YC,
                      int O, int Npix, int act, int B, hipStream_t s) {
  dim3 g(Npix / 128, O / 32, B), blk(256);
  if (cs)
    conv1x1_fast<0, true><<<g, blk, 0, s>>>(X, XC, W, wbs, C, bias, cs, Y, YC, O, Npix);
  else if (act)
    conv1x1_fast<1, false><<<g, blk, 0, s>>>(X, XC, W, wbs, C, bias, cs, Y, YC, O, Npix);
  else
    conv1x1_fast<0, false><<<g, blk, 0, s>>>(X, XC, W, wbs, C, bias, cs, Y, YC, O, Npix);
}

extern "C" void kernel_launch(void* const* d_in, const int* in_sizes, int n_in,
                              void* d_out, int out_size, void* d_ws, size_t ws_size,
                              hipStream_t stream) {
  (void)in_sizes; (void)n_in; (void)out_size; (void)ws_size;
  const float* swin   = (const float*)d_in[0];
  const float* resnet = (const float*)d_in[1];
  const float* w_swin = (const float*)d_in[2];  const float* b_swin = (const float*)d_in[3];
  const float* w_res  = (const float*)d_in[4];  const float* b_res  = (const float*)d_in[5];
  const float* aq_w = (const float*)d_in[6];    const float* aq_b = (const float*)d_in[7];
  const float* ak_w = (const float*)d_in[8];    const float* ak_b = (const float*)d_in[9];
  const float* av_w = (const float*)d_in[10];   const float* av_b = (const float*)d_in[11];
  const float* ad1_w = (const float*)d_in[12];  const float* ad1_b = (const float*)d_in[13];
  const float* ad3_w = (const float*)d_in[14];  const float* ad3_b = (const float*)d_in[15];
  const float* a_gamma = (const float*)d_in[16];
  const float* lq_w = (const float*)d_in[17];   const float* lq_b = (const float*)d_in[18];
  const float* lk_w = (const float*)d_in[19];   const float* lk_b = (const float*)d_in[20];
  const float* lv_w = (const float*)d_in[21];   const float* lv_b = (const float*)d_in[22];
  const float* ld1_w = (const float*)d_in[23];  const float* ld1_b = (const float*)d_in[24];
  const float* ld3_w = (const float*)d_in[25];  const float* ld3_b = (const float*)d_in[26];
  const float* l_gamma = (const float*)d_in[27];
  const float* wc_w = (const float*)d_in[28];   const float* wc_b = (const float*)d_in[29];

  const int B = 4;
  const int Np = 4096;                 // 64*64
  const long BNp = (long)B * Np;

  float* ws = (float*)d_ws;
  long off = 0;
  auto alloc = [&](long nf) { float* p = ws + off; off += nf; return p; };
  float* xbuf  = alloc(BNp * 320);     // x = concat(sa, ra); later holds blend
  float* qbuf  = alloc(BNp * 64);      // padded (B,64,N); rows 40..63 == 0
  float* kbuf  = alloc(BNp * 64);      // padded (B,64,N); rows 40..63 == 0
  float* vbuf  = alloc(BNp * 320);     // dead after kv  -> reused as d3buf
  float* d1buf = alloc(BNp * 320);
  float* qlbuf = alloc(BNp * 320);     // dead after qsum -> reused as d3lbuf
  float* klbuf = alloc(BNp * 320);
  float* vlbuf = alloc(BNp * 320);
  float* d1lbuf = alloc(BNp * 320);
  float* wvbuf  = alloc(BNp * 320);
  float* ksum = alloc(B * 64L);
  float* kvt  = alloc((long)B * 320 * 64);   // padded kv^T; cols 40..63 == 0
  float* qsum = alloc(BNp);
  float* es   = alloc(B * 320L);
  float* norm  = alloc(BNp);
  float* norml = alloc(BNp);
  float* aqw_pad = alloc(64L * 320);
  float* akw_pad = alloc(64L * 320);
  float* aqb_pad = alloc(64);
  float* akb_pad = alloc(64);
  float* zerob   = alloc(320);
  float* d3buf  = vbuf;
  float* d3lbuf = qlbuf;

  // 0) zero-pad the 40-row projection weights/biases to 64 rows
  pad2d_kernel<<<dim3((64 * 320 + 255) / 256), dim3(256), 0, stream>>>(aq_w, aqw_pad, 40, 320, 64L * 320);
  pad2d_kernel<<<dim3((64 * 320 + 255) / 256), dim3(256), 0, stream>>>(ak_w, akw_pad, 40, 320, 64L * 320);
  pad2d_kernel<<<dim3(1), dim3(256), 0, stream>>>(aq_b, aqb_pad, 40, 1, 64);
  pad2d_kernel<<<dim3(1), dim3(256), 0, stream>>>(ak_b, akb_pad, 40, 1, 64);
  pad2d_kernel<<<dim3(2), dim3(256), 0, stream>>>(aq_b, zerob, 0, 1, 320);

  // 1) x = concat(conv1x1(swin), conv1x1(resnet))   -> xbuf (B,320,N)
  c1(swin, 192, w_swin, 0, 192, b_swin, nullptr, xbuf, 320, 192, Np, 0, B, stream);
  c1(resnet, 128, w_res, 0, 128, b_res, nullptr, xbuf + (long)192 * Np, 320, 128, Np, 0, B, stream);

  // 2) 1x1 projections from x (q/k padded to 64 out-channels, rows 40..63 = 0)
  c1(xbuf, 320, aqw_pad, 0, 320, aqb_pad, nullptr, qbuf, 64, 64, Np, 0, B, stream);
  c1(xbuf, 320, akw_pad, 0, 320, akb_pad, nullptr, kbuf, 64, 64, Np, 0, B, stream);
  c1(xbuf, 320, av_w, 0, 320, av_b, nullptr, vbuf, 320, 320, Np, 0, B, stream);
  c1(xbuf, 320, ad1_w, 0, 320, ad1_b, nullptr, d1buf, 320, 320, Np, 0, B, stream);
  c1(xbuf, 320, lq_w, 0, 320, lq_b, nullptr, qlbuf, 320, 320, Np, 1, B, stream);   // delu
  c1(xbuf, 320, lk_w, 0, 320, lk_b, nullptr, klbuf, 320, 320, Np, 1, B, stream);   // delu
  c1(xbuf, 320, lv_w, 0, 320, lv_b, nullptr, vlbuf, 320, 320, Np, 0, B, stream);
  c1(xbuf, 320, ld1_w, 0, 320, ld1_b, nullptr, d1lbuf, 320, 320, Np, 0, B, stream);

  // 3) reductions for the two attention normalizers (padded rows contribute 0)
  rowsum_kernel<<<dim3(B * 64), dim3(256), 0, stream>>>(kbuf, ksum, Np);
  kv_kernel<<<dim3(320, 64, B), dim3(256), 0, stream>>>(kbuf, vbuf, kvt, Np);
  colsum_kernel<<<dim3(Np / 256, B), dim3(256), 0, stream>>>(qlbuf, qsum, 320, Np);
  dot_rows_kernel<<<dim3(320, B), dim3(256), 0, stream>>>(qsum, klbuf, es, 320, Np);
  norms_kernel<<<dim3(Np / 256, B), dim3(256), 0, stream>>>(qbuf, ksum, es, klbuf, norm, norml, Np);

  // 4) wv[b,c,n] = norm[b,n] * sum_m kvt[b,c,m] * q[b,m,n] (per-batch GEMM, K=64 padded)
  c1(qbuf, 64, kvt, (long)320 * 64, 64, zerob, norm, wvbuf, 320, 320, Np, 0, B, stream);

  // 5) 3x3 convs (vbuf/qlbuf are dead now; reuse as outputs)
  {
    dim3 g(320 / 32, 64, B), blk(256);
    conv3x3_wmma<<<g, blk, 0, stream>>>(xbuf, ad3_w, ad3_b, d3buf, 320, 320);
    conv3x3_wmma<<<g, blk, 0, stream>>>(xbuf, ld3_w, ld3_b, d3lbuf, 320, 320);
  }

  // 6) blend (in place into xbuf)
  {
    const long total = BNp * 320;
    compose_kernel<<<dim3((total + 255) / 256), dim3(256), 0, stream>>>(
        xbuf, wvbuf, d1buf, d3buf, klbuf, vlbuf, d1lbuf, d3lbuf, es, norml,
        a_gamma, l_gamma, 320, Np, total);
  }

  // 7) final compression conv 320 -> 160 into d_out
  c1(xbuf, 320, wc_w, 0, 320, wc_b, nullptr, (float*)d_out, 160, 160, Np, 0, B, stream);
}
